// VecSsm_70566312673392
// MI455X (gfx1250) — matrix operand order repaired and checked by reading, not compiled, not run on hardware
//
#include <hip/hip_runtime.h>
#include <math.h>

// ---------------------------------------------------------------------------
// CDNA5 WMMA types / helpers
// ---------------------------------------------------------------------------
typedef __attribute__((ext_vector_type(16))) __bf16 v16bf;
typedef __attribute__((ext_vector_type(8)))  float  v8f;
typedef __attribute__((ext_vector_type(4)))  int    v4i;

// address-space qualified pointee types for the async-DMA builtins
typedef __attribute__((address_space(1))) v4i global_v4i;
typedef __attribute__((address_space(3))) v4i lds_v4i;

union BF16x16 { v16bf v; unsigned int u[8]; };

// f32 pair -> packed bf16 dword via native converts (v_cvt_pk_bf16_f32 class)
__device__ inline unsigned int pack2bf(float f0, float f1) {
  union { __bf16 h[2]; unsigned int u; } r;
  r.h[0] = (__bf16)f0;
  r.h[1] = (__bf16)f1;
  return r.u;
}

__device__ inline v8f wmma_bf16(v16bf a, v16bf b, v8f c) {
  // D = A(16x32 bf16) * B(32x16 bf16) + C(16x16 f32)
  return __builtin_amdgcn_wmma_f32_16x16x32_bf16(
      /*neg_a=*/false, a, /*neg_b=*/false, b,
      /*c_mod=*/(short)0, c, /*reuse_a=*/false, /*reuse_b=*/false);
}

__device__ inline float gelu_f(float x) {
  return 0.5f * x * (1.0f + erff(x * 0.70710678118654752f));
}

// ---------------------------------------------------------------------------
// B fragment (32x16, K<=32 zero-padded) from global row-major W [K x ld]
// lanes 0-15 hold K=0..15 (2 per VGPR), lanes 16-31 hold K=16..31
// ---------------------------------------------------------------------------
__device__ inline v16bf make_bfrag_g(const float* W, int ld, int colBase,
                                     int K, int lane) {
  BF16x16 b;
  int n = colBase + (lane & 15);
  int klo = (lane & 16) ? 16 : 0;
#pragma unroll
  for (int j = 0; j < 8; ++j) {
    int k = klo + 2 * j;
    float f0 = (k < K)     ? W[k * ld + n]       : 0.f;
    float f1 = (k + 1 < K) ? W[(k + 1) * ld + n] : 0.f;
    b.u[j] = pack2bf(f0, f1);
  }
  return b.v;
}

// ---------------------------------------------------------------------------
// Feature transform kernel: per-token MLP stack over 1.28M tokens of dim 32.
// 8 waves / block; each wave owns 4 private LDS tiles (16 tokens x 32 feats):
//   I0/I1 : async double-buffered input (read-only during compute)
//   S, C  : scratch / intermediate
// ---------------------------------------------------------------------------
#define FT_STRIDE 34          // padded row stride (floats): bank-conflict free
#define FT_TILES_PER_WAVE 8
#define FT_BUF_FLOATS (16 * FT_STRIDE)

#ifndef FT_TRY_ASYNC
#define FT_TRY_ASYNC 1
#endif
#if FT_TRY_ASYNC && \
    __has_builtin(__builtin_amdgcn_global_load_async_to_lds_b128) && \
    __has_builtin(__builtin_amdgcn_s_wait_asynccnt)
#define FT_ASYNC 1
#else
#define FT_ASYNC 0
#endif

// A fragment (16x32 bf16, K<=kMax zero-padded) from an LDS tile
__device__ inline v16bf make_afrag_lds(const float* buf, int kMax, int lane) {
  BF16x16 a;
  int m = lane & 15;
  int hi8 = (lane & 16) ? 8 : 0;
  const float* row = buf + m * FT_STRIDE;
#pragma unroll
  for (int j = 0; j < 8; ++j) {
    int kb = 2 * hi8 + 2 * j;
    float f0 = (kb < kMax)     ? row[kb]     : 0.f;
    float f1 = (kb + 1 < kMax) ? row[kb + 1] : 0.f;
    a.u[j] = pack2bf(f0, f1);
  }
  return a.v;
}

// write a 16x16 f32 accumulator tile into an LDS tile (opt relu / residual)
__device__ inline void store_tile_lds(v8f acc, float* out, int colBase,
                                      int lane, int doRelu,
                                      const float* residBuf) {
  int col = colBase + (lane & 15);
  int rh = (lane & 16) ? 8 : 0;
#pragma unroll
  for (int r = 0; r < 8; ++r) {
    float v = acc[r];
    if (doRelu) v = fmaxf(v, 0.f);
    if (residBuf) v += residBuf[(rh + r) * FT_STRIDE + col];
    out[(rh + r) * FT_STRIDE + col] = v;
  }
}

// issue async copy of one 16-token x 32-float tile into an LDS buffer
// (4 x B128 per wave: each lane moves 16 bytes per instruction)
__device__ inline void ft_issue_tile_async(const float* __restrict__ x,
                                           long t0, float* dst, int lane) {
#if FT_ASYNC
#pragma unroll
  for (int i = 0; i < 4; ++i) {
    int e = i * 32 + lane;        // 16-byte chunk id within the tile
    int r = e >> 3;               // token row
    int c = (e & 7) * 4;          // float column
    const float* g = x + (t0 + r) * 32 + c;
    float* l = dst + r * FT_STRIDE + c;
    __builtin_amdgcn_global_load_async_to_lds_b128(
        (global_v4i*)g, (lds_v4i*)l, 0, 0);
  }
#else
  (void)x; (void)t0; (void)dst; (void)lane;
#endif
}

__global__ __launch_bounds__(256) void feature_transform_kernel(
    const float* __restrict__ x,
    const float* __restrict__ w11, const float* __restrict__ w12,
    const float* __restrict__ w13,
    const float* __restrict__ ln1g, const float* __restrict__ ln1b,
    const float* __restrict__ ftw1, const float* __restrict__ ftb1,
    const float* __restrict__ w21, const float* __restrict__ w22,
    const float* __restrict__ w23,
    const float* __restrict__ ln2g, const float* __restrict__ ln2b,
    const float* __restrict__ ftw2, const float* __restrict__ ftb2,
    float* __restrict__ hbuf) {
  extern __shared__ float dynlds[];   // 8 waves * 4 bufs * 16*34 floats

  const int lane = threadIdx.x & 31;
  const int wave = threadIdx.x >> 5;
  const int hi = lane >> 4;

  float* wbase = dynlds + wave * 4 * FT_BUF_FLOATS;
  float* I0 = wbase;
  float* I1 = wbase + FT_BUF_FLOATS;
  float* S  = wbase + 2 * FT_BUF_FLOATS;
  float* C  = wbase + 3 * FT_BUF_FLOATS;

  // --- preload all weight B-fragments (held in VGPRs across the tile loop)
  v16bf B11a = make_bfrag_g(w11, 32, 0, 32, lane);
  v16bf B11b = make_bfrag_g(w11, 32, 16, 32, lane);
  v16bf B12a = make_bfrag_g(w12, 32, 0, 32, lane);
  v16bf B12b = make_bfrag_g(w12, 32, 16, 32, lane);
  v16bf B13a = make_bfrag_g(w13, 32, 0, 32, lane);
  v16bf B13b = make_bfrag_g(w13, 32, 16, 32, lane);
  v16bf Bf1  = make_bfrag_g(ftw1, 16, 0, 32, lane);
  v16bf B21  = make_bfrag_g(w21, 16, 0, 16, lane);
  v16bf B22  = make_bfrag_g(w22, 16, 0, 16, lane);
  v16bf B23  = make_bfrag_g(w23, 16, 0, 16, lane);

  const long waveBase =
      ((long)blockIdx.x * 8 + wave) * (FT_TILES_PER_WAVE * 16L);

#if FT_ASYNC
  ft_issue_tile_async(x, waveBase, I0, lane);   // prefetch tile 0
#endif

  for (int tile = 0; tile < FT_TILES_PER_WAVE; ++tile) {
    const long t0 = waveBase + tile * 16L;
    float* in  = (tile & 1) ? I1 : I0;
    float* nxt = (tile & 1) ? I0 : I1;

#if FT_ASYNC
    // overlap next tile's global->LDS copy with this tile's WMMA compute
    if (tile < FT_TILES_PER_WAVE - 1) {
      ft_issue_tile_async(x, t0 + 16, nxt, lane);
      __builtin_amdgcn_s_wait_asynccnt(4);   // tile's own 4 copies retired
    } else {
      __builtin_amdgcn_s_wait_asynccnt(0);
    }
#else
    // synchronous copy of this tile + prefetch of the next one
    {
      int r = lane >> 1;
      int c = (lane & 1) * 16;
      const float* src = x + (t0 + r) * 32 + c;
      float* dst = in + r * FT_STRIDE + c;
#pragma unroll
      for (int i = 0; i < 16; i += 2) {
        float2 v = *(const float2*)(src + i);
        *(float2*)(dst + i) = v;
      }
    }
    if (tile + 1 < FT_TILES_PER_WAVE)
      __builtin_prefetch(x + (t0 + 16) * 32 + lane * 16, 0, 1);
    (void)nxt;
#endif
    __syncthreads();

    // ---- res_block1: h = x + relu(relu(x@w1)@w2)@w3   ('in' is read-only)
    {
      v16bf a = make_afrag_lds(in, 32, lane);
      v8f c0 = {}, c1 = {};
      c0 = wmma_bf16(a, B11a, c0);
      c1 = wmma_bf16(a, B11b, c1);
      store_tile_lds(c0, S, 0, lane, 1, nullptr);
      store_tile_lds(c1, S, 16, lane, 1, nullptr);
    }
    __syncthreads();
    {
      v16bf a = make_afrag_lds(S, 32, lane);
      v8f c0 = {}, c1 = {};
      c0 = wmma_bf16(a, B12a, c0);
      c1 = wmma_bf16(a, B12b, c1);
      store_tile_lds(c0, S, 0, lane, 1, nullptr);     // in-place, same wave
      store_tile_lds(c1, S, 16, lane, 1, nullptr);
    }
    __syncthreads();
    {
      v16bf a = make_afrag_lds(S, 32, lane);
      v8f c0 = {}, c1 = {};
      c0 = wmma_bf16(a, B13a, c0);
      c1 = wmma_bf16(a, B13b, c1);
      store_tile_lds(c0, C, 0, lane, 0, in);          // h = x + (.)  -> C
      store_tile_lds(c1, C, 16, lane, 0, in);
    }
    __syncthreads();

    // ---- LayerNorm over 32 features (2 lanes per token via shfl_xor 16)
    {
      int t = lane & 15;
      float* row = C + t * FT_STRIDE + hi * 16;
      float v[16];
      float s = 0.f, ss = 0.f;
#pragma unroll
      for (int j = 0; j < 16; ++j) {
        v[j] = row[j];
        s += v[j];
        ss += v[j] * v[j];
      }
      s += __shfl_xor(s, 16, 32);
      ss += __shfl_xor(ss, 16, 32);
      float mu = s * (1.f / 32.f);
      float var = ss * (1.f / 32.f) - mu * mu;
      float rstd = rsqrtf(var + 1e-5f);
#pragma unroll
      for (int j = 0; j < 16; ++j) {
        int fc = hi * 16 + j;
        row[j] = (v[j] - mu) * rstd * ln1g[fc] + ln1b[fc];
      }
    }
    __syncthreads();

    // ---- ft_w1 (32 -> 16) + bias + relu -> S cols 0-15
    {
      v16bf a = make_afrag_lds(C, 32, lane);
      v8f c0 = {};
      c0 = wmma_bf16(a, Bf1, c0);
      int col = lane & 15;
      int rh = (lane & 16) ? 8 : 0;
      float bb = ftb1[col];
#pragma unroll
      for (int r = 0; r < 8; ++r)
        S[(rh + r) * FT_STRIDE + col] = fmaxf(c0[r] + bb, 0.f);
    }
    __syncthreads();

    // ---- res_block2 (16x16, K padded to 32 with zeros)
    {
      v16bf a = make_afrag_lds(S, 16, lane);
      v8f c0 = {};
      c0 = wmma_bf16(a, B21, c0);
      store_tile_lds(c0, C, 0, lane, 1, nullptr);
    }
    __syncthreads();
    {
      v16bf a = make_afrag_lds(C, 16, lane);
      v8f c0 = {};
      c0 = wmma_bf16(a, B22, c0);
      store_tile_lds(c0, C, 0, lane, 1, nullptr);     // in-place, same wave
    }
    __syncthreads();
    {
      v16bf a = make_afrag_lds(C, 16, lane);
      v8f c0 = {};
      c0 = wmma_bf16(a, B23, c0);
      store_tile_lds(c0, S, 0, lane, 0, S);           // + residual
    }
    __syncthreads();

    // ---- LayerNorm over 16 features (lanes 0-15, one token each)
    if (lane < 16) {
      float* row = S + lane * FT_STRIDE;
      float v[16];
      float s = 0.f, ss = 0.f;
#pragma unroll
      for (int j = 0; j < 16; ++j) {
        v[j] = row[j];
        s += v[j];
        ss += v[j] * v[j];
      }
      float mu = s * (1.f / 16.f);
      float var = ss * (1.f / 16.f) - mu * mu;
      float rstd = rsqrtf(var + 1e-5f);
#pragma unroll
      for (int j = 0; j < 16; ++j)
        row[j] = (v[j] - mu) * rstd * ln2g[j] + ln2b[j];
    }
    __syncthreads();

    // ---- ft_w2 (16 -> 1) + bias + relu -> hbuf[token]
    if (lane < 16) {
      const float* row = S + lane * FT_STRIDE;
      float s = ftb2[0];
#pragma unroll
      for (int j = 0; j < 16; ++j) s += row[j] * ftw2[j];
      hbuf[t0 + lane] = fmaxf(s, 0.f);
    }
    __syncthreads();
  }
}

// ---------------------------------------------------------------------------
// Generic WMMA bf16 GEMM: C = epilogue(A[MxK] @ W' + bias) (+ residual)
//   W' = W[KxN] row-major (transW=0) or W[NxK] row-major accessed as (k,n)
//   act: 0 none, 1 relu, 2 gelu;  wscale applied to W on convert (e.g. -1)
// One wave per 16x16 output tile. M must be a multiple of 16; N guarded.
// ---------------------------------------------------------------------------
__global__ __launch_bounds__(32) void wmma_gemm_kernel(
    const float* __restrict__ A, int lda,
    const float* __restrict__ W, int ldw, int transW,
    const float* __restrict__ bias,
    const float* __restrict__ resid, int ldr,
    float* __restrict__ C, int ldc,
    int M, int N, int K, int act, float wscale) {
  const int lane = threadIdx.x & 31;
  const int rowBase = blockIdx.x * 16;
  const int colBase = blockIdx.y * 16;
  const int mrow = rowBase + (lane & 15);
  const int hi8 = (lane & 16) ? 8 : 0;
  const int ncol = colBase + (lane & 15);
  const bool colOK = (ncol < N);
  const int klo = (lane & 16) ? 16 : 0;
  const float* arow = A + (long)mrow * lda;

  v8f acc = {};
  for (int k0 = 0; k0 < K; k0 += 32) {
    BF16x16 af, bf;
#pragma unroll
    for (int j = 0; j < 8; ++j) {
      int kb = k0 + 2 * hi8 + 2 * j;
      float f0 = (kb < K)     ? arow[kb]     : 0.f;
      float f1 = (kb + 1 < K) ? arow[kb + 1] : 0.f;
      af.u[j] = pack2bf(f0, f1);
    }
#pragma unroll
    for (int j = 0; j < 8; ++j) {
      int k = k0 + klo + 2 * j;
      float f0 = 0.f, f1 = 0.f;
      if (colOK) {
        if (transW) {
          const float* wr = W + (long)ncol * ldw;
          f0 = (k < K)     ? wr[k]     : 0.f;
          f1 = (k + 1 < K) ? wr[k + 1] : 0.f;
        } else {
          f0 = (k < K)     ? W[(long)k * ldw + ncol]       : 0.f;
          f1 = (k + 1 < K) ? W[(long)(k + 1) * ldw + ncol] : 0.f;
        }
      }
      bf.u[j] = pack2bf(f0 * wscale, f1 * wscale);
    }
    acc = wmma_bf16(af.v, bf.v, acc);
  }

  if (colOK) {
    float bv = bias ? bias[ncol] : 0.f;
#pragma unroll
    for (int r = 0; r < 8; ++r) {
      int row = rowBase + hi8 + r;
      float v = acc[r] + bv;
      if (act == 1) v = fmaxf(v, 0.f);
      else if (act == 2) v = gelu_f(v);
      if (resid) v += resid[(long)row * ldr + ncol];
      C[(long)row * ldc + ncol] = v;
    }
  }
}

// ---------------------------------------------------------------------------
// Row LayerNorm: one block (256 threads) per row of X[rows x F]
// ---------------------------------------------------------------------------
__global__ __launch_bounds__(256) void ln_rows_kernel(
    const float* __restrict__ X, const float* __restrict__ g,
    const float* __restrict__ b, float* __restrict__ Y, int F) {
  __shared__ float red[18];
  long row = blockIdx.x;
  const float* xr = X + row * F;
  float* yr = Y + row * F;
  float s = 0.f, ss = 0.f;
  for (int i = threadIdx.x; i < F; i += 256) {
    float v = xr[i];
    s += v;
    ss += v * v;
  }
#pragma unroll
  for (int o = 16; o > 0; o >>= 1) {
    s += __shfl_xor(s, o, 32);
    ss += __shfl_xor(ss, o, 32);
  }
  int wid = threadIdx.x >> 5;
  if ((threadIdx.x & 31) == 0) { red[wid] = s; red[8 + wid] = ss; }
  __syncthreads();
  if (threadIdx.x == 0) {
    float S = 0.f, SS = 0.f;
    for (int w = 0; w < 8; ++w) { S += red[w]; SS += red[8 + w]; }
    red[16] = S; red[17] = SS;
  }
  __syncthreads();
  float mu = red[16] / (float)F;
  float var = red[17] / (float)F - mu * mu;
  float rstd = rsqrtf(var + 1e-5f);
  for (int i = threadIdx.x; i < F; i += 256)
    yr[i] = (xr[i] - mu) * rstd * g[i] + b[i];
}

// ---------------------------------------------------------------------------
// S5 discretization: lam_bar = exp(lam*dt), B_bar = ((lam_bar-1)/lam) * Bc
// grid: P=256 blocks, H=256 threads
// ---------------------------------------------------------------------------
__global__ __launch_bounds__(256) void precompute_ssm_kernel(
    const float* __restrict__ lam_re, const float* __restrict__ lam_im,
    const float* __restrict__ B_re, const float* __restrict__ B_im,
    const float* __restrict__ log_step,
    float* __restrict__ Bb_re, float* __restrict__ Bb_im,
    float* __restrict__ lamb_re, float* __restrict__ lamb_im) {
  int p = blockIdx.x, h = threadIdx.x;
  float lr = lam_re[p], li = lam_im[p];
  float dt = expf(log_step[p]);
  float mag = expf(lr * dt);
  float lbr = mag * cosf(li * dt);
  float lbi = mag * sinf(li * dt);
  float nr = lbr - 1.0f, ni = lbi;
  float den = lr * lr + li * li;
  float cr = (nr * lr + ni * li) / den;
  float ci = (ni * lr - nr * li) / den;
  int idx = p * 256 + h;
  float br = B_re[idx], bi = B_im[idx];
  Bb_re[idx] = cr * br - ci * bi;
  Bb_im[idx] = cr * bi + ci * br;
  if (h == 0) { lamb_re[p] = lbr; lamb_im[p] = lbi; }
}

// ---------------------------------------------------------------------------
// Sequential complex scan over L=64 per (b,p) chain (B=4 blocks, P=256 thr)
// ---------------------------------------------------------------------------
__global__ __launch_bounds__(256) void ssm_scan_kernel(
    const float* __restrict__ bu_re, const float* __restrict__ bu_im,
    const float* __restrict__ lamb_re, const float* __restrict__ lamb_im,
    float* __restrict__ xs_re, float* __restrict__ xs_im) {
  int b = blockIdx.x, p = threadIdx.x;
  float lr = lamb_re[p], li = lamb_im[p];
  float sr = 0.f, si = 0.f;
  for (int l = 0; l < 64; ++l) {
    long idx = ((long)(b * 64 + l)) * 256 + p;
    float br = bu_re[idx], bi = bu_im[idx];
    float nr = lr * sr - li * si + br;
    float ni = lr * si + li * sr + bi;
    sr = nr; si = ni;
    xs_re[idx] = sr;
    xs_im[idx] = si;
  }
}

// z = gelu(yre + Dp*fx) + fx
__global__ __launch_bounds__(256) void ssm_post_kernel(
    const float* __restrict__ yre, const float* __restrict__ fx,
    const float* __restrict__ Dp, float* __restrict__ z) {
  long idx = (long)blockIdx.x * 256 + threadIdx.x;
  float y = yre[idx] + Dp[threadIdx.x] * fx[idx];
  z[idx] = gelu_f(y) + fx[idx];
}

// GEGLU combine: gm = a * gelu(b), a = gg[:, :256], b = gg[:, 256:]
__global__ __launch_bounds__(256) void geglu_kernel(
    const float* __restrict__ gg, float* __restrict__ gm) {
  long row = blockIdx.x;
  int j = threadIdx.x;
  float a = gg[row * 512 + j];
  float b = gg[row * 512 + 256 + j];
  gm[row * 256 + j] = a * gelu_f(b);
}

// ---------------------------------------------------------------------------
// Host-side launcher
// ---------------------------------------------------------------------------
extern "C" void kernel_launch(void* const* d_in, const int* in_sizes, int n_in,
                              void* d_out, int out_size, void* d_ws,
                              size_t ws_size, hipStream_t stream) {
  (void)in_sizes; (void)n_in; (void)out_size; (void)ws_size;
  // setup_inputs() dict order
  const float* x        = (const float*)d_in[0];
  const float* rb1_w1   = (const float*)d_in[1];
  const float* rb2_w1   = (const float*)d_in[2];
  const float* rb1_w2   = (const float*)d_in[3];
  const float* rb2_w2   = (const float*)d_in[4];
  const float* rb1_w3   = (const float*)d_in[5];
  const float* rb2_w3   = (const float*)d_in[6];
  const float* ln1_g    = (const float*)d_in[7];
  const float* ln1_b    = (const float*)d_in[8];
  const float* ft_w1    = (const float*)d_in[9];
  const float* ft_b1    = (const float*)d_in[10];
  const float* ln2_g    = (const float*)d_in[11];
  const float* ln2_b    = (const float*)d_in[12];
  const float* ft_w2    = (const float*)d_in[13];
  const float* ft_b2    = (const float*)d_in[14];
  const float* br_w1    = (const float*)d_in[15];
  const float* br_b1    = (const float*)d_in[16];
  const float* br_ln_g  = (const float*)d_in[17];
  const float* br_ln_b  = (const float*)d_in[18];
  const float* br_w2    = (const float*)d_in[19];
  const float* br_b2    = (const float*)d_in[20];
  const float* attn_ln_g = (const float*)d_in[21];
  const float* attn_ln_b = (const float*)d_in[22];
  const float* lam_re   = (const float*)d_in[23];
  const float* lam_im   = (const float*)d_in[24];
  const float* B_re     = (const float*)d_in[25];
  const float* B_im     = (const float*)d_in[26];
  const float* C_re     = (const float*)d_in[27];
  const float* C_im     = (const float*)d_in[28];
  const float* Dp       = (const float*)d_in[29];
  const float* log_step = (const float*)d_in[30];
  const float* ff_ln_g  = (const float*)d_in[31];
  const float* ff_ln_b  = (const float*)d_in[32];
  const float* ff_enc_w = (const float*)d_in[33];
  const float* ff_dec_w = (const float*)d_in[34];
  const float* hd_w1    = (const float*)d_in[35];
  const float* hd_b1    = (const float*)d_in[36];
  const float* hd_w2    = (const float*)d_in[37];
  const float* hd_b2    = (const float*)d_in[38];
  const float* hd_w3    = (const float*)d_in[39];
  const float* hd_b3    = (const float*)d_in[40];
  float* out = (float*)d_out;

  // workspace carve-up (floats)
  float* ws = (float*)d_ws;
  float* hbuf   = ws; ws += 1280000;       // [256 x 5000]
  float* g1     = ws; ws += 256 * 512;
  float* g1n    = ws; ws += 256 * 512;
  float* ubuf   = ws; ws += 256 * 256;
  float* fx     = ws; ws += 256 * 256;
  float* Bb_re  = ws; ws += 256 * 256;
  float* Bb_im  = ws; ws += 256 * 256;
  float* lamb_re = ws; ws += 256;
  float* lamb_im = ws; ws += 256;
  float* bu_re  = ws; ws += 256 * 256;
  float* bu_im  = ws; ws += 256 * 256;
  float* xs_re  = ws; ws += 256 * 256;
  float* xs_im  = ws; ws += 256 * 256;
  float* ybuf   = ws; ws += 256 * 256;
  float* zbuf   = ws; ws += 256 * 256;
  float* fx2    = ws; ws += 256 * 256;
  float* gg     = ws; ws += 256 * 512;
  float* gm     = ws; ws += 256 * 256;
  float* ff     = ws; ws += 256 * 256;
  float* o1     = ws; ws += 256 * 128;
  float* o2     = ws; ws += 256 * 128;

  // 1) feature transform over all 1.28M tokens (async double-buffered LDS)
  const size_t ft_lds_bytes = 8u * 4u * FT_BUF_FLOATS * sizeof(float); // 69632
  feature_transform_kernel<<<1250, 256, ft_lds_bytes, stream>>>(
      x, rb1_w1, rb1_w2, rb1_w3, ln1_g, ln1_b, ft_w1, ft_b1,
      rb2_w1, rb2_w2, rb2_w3, ln2_g, ln2_b, ft_w2, ft_b2, hbuf);

  // 2) bridge: relu(h @ br_w1 + b), LN, @ br_w2 + b, LN
  wmma_gemm_kernel<<<dim3(16, 32), 32, 0, stream>>>(
      hbuf, 5000, br_w1, 512, 0, br_b1, nullptr, 0, g1, 512,
      256, 512, 5000, /*act=*/1, 1.0f);
  ln_rows_kernel<<<256, 256, 0, stream>>>(g1, br_ln_g, br_ln_b, g1n, 512);
  wmma_gemm_kernel<<<dim3(16, 16), 32, 0, stream>>>(
      g1n, 512, br_w2, 256, 0, br_b2, nullptr, 0, ubuf, 256,
      256, 256, 512, 0, 1.0f);
  ln_rows_kernel<<<256, 256, 0, stream>>>(ubuf, attn_ln_g, attn_ln_b, fx, 256);

  // 3) S5 block
  precompute_ssm_kernel<<<256, 256, 0, stream>>>(
      lam_re, lam_im, B_re, B_im, log_step, Bb_re, Bb_im, lamb_re, lamb_im);
  // bu = fx @ B_bar^T (re / im)
  wmma_gemm_kernel<<<dim3(16, 16), 32, 0, stream>>>(
      fx, 256, Bb_re, 256, /*trans=*/1, nullptr, nullptr, 0, bu_re, 256,
      256, 256, 256, 0, 1.0f);
  wmma_gemm_kernel<<<dim3(16, 16), 32, 0, stream>>>(
      fx, 256, Bb_im, 256, 1, nullptr, nullptr, 0, bu_im, 256,
      256, 256, 256, 0, 1.0f);
  ssm_scan_kernel<<<4, 256, 0, stream>>>(bu_re, bu_im, lamb_re, lamb_im,
                                         xs_re, xs_im);
  // y = Re(xs @ C^T) = xs_re @ C_re^T - xs_im @ C_im^T
  wmma_gemm_kernel<<<dim3(16, 16), 32, 0, stream>>>(
      xs_re, 256, C_re, 256, 1, nullptr, nullptr, 0, ybuf, 256,
      256, 256, 256, 0, 1.0f);
  wmma_gemm_kernel<<<dim3(16, 16), 32, 0, stream>>>(
      xs_im, 256, C_im, 256, 1, nullptr, /*resid=*/ybuf, 256, ybuf, 256,
      256, 256, 256, 0, /*wscale=*/-1.0f);
  ssm_post_kernel<<<256, 256, 0, stream>>>(ybuf, fx, Dp, zbuf);
  ln_rows_kernel<<<256, 256, 0, stream>>>(zbuf, ff_ln_g, ff_ln_b, fx2, 256);
  // GEGLU FFN
  wmma_gemm_kernel<<<dim3(16, 32), 32, 0, stream>>>(
      fx2, 256, ff_enc_w, 512, 0, nullptr, nullptr, 0, gg, 512,
      256, 512, 256, 0, 1.0f);
  geglu_kernel<<<256, 256, 0, stream>>>(gg, gm);
  wmma_gemm_kernel<<<dim3(16, 16), 32, 0, stream>>>(
      gm, 256, ff_dec_w, 256, 0, nullptr, /*resid=*/fx2, 256, ff, 256,
      256, 256, 256, 0, 1.0f);

  // 4) classification head
  wmma_gemm_kernel<<<dim3(16, 8), 32, 0, stream>>>(
      ff, 256, hd_w1, 128, 0, hd_b1, nullptr, 0, o1, 128,
      256, 128, 256, 1, 1.0f);
  wmma_gemm_kernel<<<dim3(16, 8), 32, 0, stream>>>(
      o1, 128, hd_w2, 128, 0, hd_b2, nullptr, 0, o2, 128,
      256, 128, 128, 1, 1.0f);
  wmma_gemm_kernel<<<dim3(16, 7), 32, 0, stream>>>(
      o2, 128, hd_w3, 100, 0, hd_b3, nullptr, 0, out, 100,
      256, 100, 128, 0, 1.0f);
}
